// WaveNetLayer_39058432590033
// MI455X (gfx1250) — compile-verified, matrix operand order checked by reading, not run on hardware
//
#include <hip/hip_runtime.h>

// ---------------------------------------------------------------------------
// WaveNet stack on MI455X (gfx1250, wave32).
// All matmuls via v_wmma_f32_16x16x32_bf16 (bf16 operands, f32 accumulate).
// Weight K-slices are staged into LDS via the CDNA5 async global->LDS path
// (double-buffered, s_wait_asynccnt + barrier), de-duplicating B-fragment
// traffic 4x across the row-stripe waves. Residual chain + skip kept in f32.
// ---------------------------------------------------------------------------

#define B_  8
#define T_  4096
#define C_  64
#define F_  256
#define BT_ (B_ * T_)   /* 32768 */

#define BSTRIDE 40      /* LDS row stride (halves) for staged weights: 80B   */

typedef __attribute__((ext_vector_type(16))) __bf16 v16bf;
typedef __attribute__((ext_vector_type(8)))  __bf16 v8bf;
typedef __attribute__((ext_vector_type(8)))  float  v8f;
typedef unsigned short u16;
typedef int vi4 __attribute__((vector_size(16)));   // builtin's pointee type

#define AS1 __attribute__((address_space(1)))
#define AS3 __attribute__((address_space(3)))

#if defined(__HIP_DEVICE_COMPILE__) && __has_builtin(__builtin_amdgcn_global_load_async_to_lds_b128)
#define HAVE_ASYNC_LDS 1
#else
#define HAVE_ASYNC_LDS 0
#endif

// ---- helpers --------------------------------------------------------------

static __device__ __forceinline__ u16 f32_to_bf16_bits(float f) {
  union { float f; unsigned int u; } v; v.f = f;
  unsigned int u = v.u;
  u += 0x7fffu + ((u >> 16) & 1u);           // round-to-nearest-even
  return (u16)(u >> 16);
}

static __device__ __forceinline__ v8bf load8bf(const u16* p) {
  union { uint4 q; v8bf v; } u;
  u.q = *(const uint4*)p;                    // 128-bit load
  return u.v;
}

static __device__ __forceinline__ v16bf combine16(v8bf lo, v8bf hi) {
  v16bf a;
#pragma unroll
  for (int i = 0; i < 8; ++i) { a[i] = lo[i]; a[i + 8] = hi[i]; }
  return a;
}

// A/B fragment load for 16x16x32 bf16 WMMA from a row-major bf16 matrix.
//   lane<16 : K[k0..k0+7]    elems 0..7, K[k0+16..k0+23] elems 8..15
//   lane>=16: K[k0+8..k0+15]             K[k0+24..k0+31]   (kk = 8)
static __device__ __forceinline__ v16bf frag_ld(const u16* row, int k0, int kk) {
  return combine16(load8bf(row + k0 + kk), load8bf(row + k0 + 16 + kk));
}

static __device__ __forceinline__ v16bf frag_zero() {
  v16bf a;
#pragma unroll
  for (int i = 0; i < 16; ++i) a[i] = (__bf16)0.0f;
  return a;
}

static __device__ __forceinline__ v8f acc_zero() {
  v8f z;
#pragma unroll
  for (int i = 0; i < 8; ++i) z[i] = 0.0f;
  return z;
}

static __device__ __forceinline__ v8f wmma_bf16(v16bf a, v16bf b, v8f c) {
  return __builtin_amdgcn_wmma_f32_16x16x32_bf16(false, a, false, b,
                                                 (short)0, c, false, false);
}

static __device__ __forceinline__ float selu_f(float x) {
  const float alpha = 1.6732632423543772f, scale = 1.0507009873554805f;
  return scale * (x > 0.0f ? x : alpha * (__expf(x) - 1.0f));
}

static __device__ __forceinline__ v8bf selu8_bf(const float* p) {
  v8bf r;
#pragma unroll
  for (int i = 0; i < 8; ++i) {
    union { u16 u; __bf16 b; } c;
    c.u = f32_to_bf16_bits(selu_f(p[i]));
    r[i] = c.b;
  }
  return r;
}

// ---- CDNA5 async global->LDS copy (16B per lane per issue) ----------------

static __device__ __forceinline__ void async_cp16(const u16* g, u16* l) {
#if HAVE_ASYNC_LDS
  __builtin_amdgcn_global_load_async_to_lds_b128(
      (AS1 vi4*)(g), (AS3 vi4*)(l), 0, 0);
#else
  *(uint4*)l = *(const uint4*)g;             // synchronous fallback
#endif
}

static __device__ __forceinline__ void wait_async8() {
#if HAVE_ASYNC_LDS
  asm volatile("s_wait_asynccnt 0x8" ::: "memory");
#endif
}
static __device__ __forceinline__ void wait_async0() {
#if HAVE_ASYNC_LDS
  asm volatile("s_wait_asynccnt 0x0" ::: "memory");
#endif
}

// Stage one 32-wide K-slice of two [256][gstride] weight matrices into LDS.
// Thread tid owns row n = tid: 4 chunks x 16B per matrix -> 8 async issues
// per wave (all lanes participate per issue).
static __device__ __forceinline__ void stage_issue_pair(
    const u16* __restrict__ g0, const u16* __restrict__ g1, int gstride,
    int kc, u16* l0, u16* l1, int tid)
{
  const int n = tid;
  const u16* gf = g0 + (size_t)n * gstride + kc * 32;
  const u16* gg = g1 + (size_t)n * gstride + kc * 32;
  u16* lf = l0 + n * BSTRIDE;
  u16* lg = l1 + n * BSTRIDE;
#pragma unroll
  for (int c = 0; c < 4; ++c) {
    async_cp16(gf + c * 8, lf + c * 8);
    async_cp16(gg + c * 8, lg + c * 8);
  }
}

// ---- prep kernels: bf16 convert + weight transposition --------------------

__global__ void cvt_bf16_kernel(const float* __restrict__ s,
                                u16* __restrict__ d, int n) {
  int i = blockIdx.x * blockDim.x + threadIdx.x;
  if (i < n) d[i] = f32_to_bf16_bits(s[i]);
}

// Wc (2,64,256) -> WcT [256 out][128 k]  (k = tap*64 + cin)
__global__ void pack_wc_kernel(const float* __restrict__ W, u16* __restrict__ WT) {
  int i = blockIdx.x * blockDim.x + threadIdx.x;
  if (i >= 256 * 128) return;
  int o = i >> 7, k = i & 127, tap = k >> 6, c = k & 63;
  WT[i] = f32_to_bf16_bits(W[(tap * 64 + c) * 256 + o]);
}

// Wf/Wg (8,2,256,256) -> [8][256 out][512 k]  (k = tap*256 + fin)
__global__ void pack_wfg_kernel(const float* __restrict__ W, u16* __restrict__ WT) {
  int i = blockIdx.x * blockDim.x + threadIdx.x;
  if (i >= 8 * 256 * 512) return;
  int l = i >> 17, r = i & 131071;
  int o = r >> 9, k = r & 511, tap = k >> 8, f = k & 255;
  WT[i] = f32_to_bf16_bits(W[(((l * 2 + tap) * 256 + f) << 8) + o]);
}

// Wr/Ws (8,256,256) -> [8][256 out][256 k]
__global__ void pack_wrs_kernel(const float* __restrict__ W, u16* __restrict__ WT) {
  int i = blockIdx.x * blockDim.x + threadIdx.x;
  if (i >= 8 * 256 * 256) return;
  int l = i >> 16, r = i & 65535, o = r >> 8, k = r & 255;
  WT[i] = f32_to_bf16_bits(W[(l << 16) + (k << 8) + o]);
}

// Wd1 (256,64) -> [64 out][256 k]
__global__ void pack_wd1_kernel(const float* __restrict__ W, u16* __restrict__ WT) {
  int i = blockIdx.x * blockDim.x + threadIdx.x;
  if (i >= 64 * 256) return;
  int c = i >> 8, f = i & 255;
  WT[i] = f32_to_bf16_bits(W[f * 64 + c]);
}

// Wd2 (64,64) -> [64 out][64 k]
__global__ void pack_wd2_kernel(const float* __restrict__ W, u16* __restrict__ WT) {
  int i = blockIdx.x * blockDim.x + threadIdx.x;
  if (i >= 64 * 64) return;
  int o = i >> 6, c = i & 63;
  WT[i] = f32_to_bf16_bits(W[c * 64 + o]);
}

// ---- initial causal conv: out = [x[t-1], x[t]] @ WcT^T ---------------------

__global__ __launch_bounds__(256) void conv_in_kernel(
    const u16* __restrict__ x_bf,       // [BT][64]
    const u16* __restrict__ WcT,        // [256][128]
    float* __restrict__ out_f32,        // [BT][256]
    u16*   __restrict__ out_bf,         // [BT][256]
    float* __restrict__ skip)           // [BT][256]
{
  const int tid = threadIdx.x;
  const int w = tid >> 5, lane = tid & 31;
  const int lrow = lane & 15, khalf = lane >> 4, kk = khalf * 8;
  const int m0 = (w >> 1) * 16;
  const int colBase = (w & 1) * 128;
  const int tile0 = blockIdx.x * 64;
  const int rowG = tile0 + m0 + lrow;
  const int t = rowG & (T_ - 1);
  const bool v0 = (t >= 1);
  const u16* a1p = x_bf + (size_t)rowG * C_;
  const u16* a0p = x_bf + (size_t)(v0 ? rowG - 1 : rowG) * C_;

  v8f acc[8];
#pragma unroll
  for (int ct = 0; ct < 8; ++ct) acc[ct] = acc_zero();

#pragma unroll
  for (int kc = 0; kc < 4; ++kc) {                 // K = 128
    const int tap = kc >> 1;
    const int k0 = (kc & 1) * 32;
    v16bf a = (tap == 0) ? (v0 ? frag_ld(a0p, k0, kk) : frag_zero())
                         : frag_ld(a1p, k0, kk);
    const int kb = kc * 32;
#pragma unroll
    for (int ct = 0; ct < 8; ++ct) {
      const int n = colBase + ct * 16 + lrow;
      acc[ct] = wmma_bf16(a, frag_ld(WcT + (size_t)n * 128, kb, kk), acc[ct]);
    }
  }

#pragma unroll
  for (int ct = 0; ct < 8; ++ct) {
    const int n = colBase + ct * 16 + lrow;
#pragma unroll
    for (int j = 0; j < 8; ++j) {
      const int m = m0 + khalf * 8 + j;
      const size_t idx = (size_t)(tile0 + m) * F_ + n;
      const float v = acc[ct][j];
      out_f32[idx] = v;
      out_bf[idx]  = f32_to_bf16_bits(v);
      skip[idx]    = 0.0f;
    }
  }
}

// ---- one residual layer ----------------------------------------------------
// f = [out[t-d],out[t]] @ WfT^T ; g likewise ; z = tanh(f)*sigmoid(g)
// out_new = out + z @ WrT^T + br ; skip += z @ WsT^T + bs

__global__ __launch_bounds__(256) void layer_kernel(
    const float* __restrict__ out_in_f32,
    const u16*   __restrict__ out_in_bf,
    float* __restrict__ out_out_f32,
    u16*   __restrict__ out_out_bf,
    float* __restrict__ skip,
    const u16* __restrict__ WfT,        // [256][512]
    const u16* __restrict__ WgT,        // [256][512]
    const u16* __restrict__ WrT,        // [256][256]
    const u16* __restrict__ WsT,        // [256][256]
    const float* __restrict__ br,       // [256]
    const float* __restrict__ bs,       // [256]
    int d)
{
  __shared__ u16 zlds[64 * 264];                  // z tile (padded)
  __shared__ u16 bstage[2][2][256 * BSTRIDE];     // [slot][f/g or r/s][slice]

  const int tid = threadIdx.x;
  const int w = tid >> 5, lane = tid & 31;
  const int lrow = lane & 15, khalf = lane >> 4, kk = khalf * 8;
  const int m0 = (w >> 1) * 16;
  const int colBase = (w & 1) * 128;
  const int tile0 = blockIdx.x * 64;
  const int rowG = tile0 + m0 + lrow;
  const int t = rowG & (T_ - 1);
  const bool v0 = (t - d) >= 0;
  const u16* a1p = out_in_bf + (size_t)rowG * F_;
  const u16* a0p = out_in_bf + (size_t)(v0 ? rowG - d : rowG) * F_;

  // ---- stage 1: f and g GEMMs, K = 512 (two taps), LDS-staged weights ----
  v8f accf[8], accg[8];
#pragma unroll
  for (int ct = 0; ct < 8; ++ct) { accf[ct] = acc_zero(); accg[ct] = acc_zero(); }

  stage_issue_pair(WfT, WgT, 512, 0, &bstage[0][0][0], &bstage[0][1][0], tid);
#pragma unroll
  for (int kc = 0; kc < 16; ++kc) {
    const int s = kc & 1;
    if (kc < 15)
      stage_issue_pair(WfT, WgT, 512, kc + 1,
                       &bstage[s ^ 1][0][0], &bstage[s ^ 1][1][0], tid);
    if (kc < 15) wait_async8(); else wait_async0();
    __syncthreads();                              // staged slice visible

    const int tap = kc >> 3;
    const int k0 = (kc & 7) * 32;
    v16bf a = (tap == 0) ? (v0 ? frag_ld(a0p, k0, kk) : frag_zero())
                         : frag_ld(a1p, k0, kk);
#pragma unroll
    for (int ct = 0; ct < 8; ++ct) {
      const int n = colBase + ct * 16 + lrow;
      accf[ct] = wmma_bf16(a, frag_ld(&bstage[s][0][(size_t)n * BSTRIDE], 0, kk), accf[ct]);
      accg[ct] = wmma_bf16(a, frag_ld(&bstage[s][1][(size_t)n * BSTRIDE], 0, kk), accg[ct]);
    }
    __syncthreads();                              // slot may be re-issued
  }

  // ---- gated activation -> bf16 z tile in LDS ----
#pragma unroll
  for (int ct = 0; ct < 8; ++ct) {
    const int n = colBase + ct * 16 + lrow;
#pragma unroll
    for (int j = 0; j < 8; ++j) {
      const int m = m0 + khalf * 8 + j;
      const float f = accf[ct][j], g = accg[ct][j];
      const float z = tanhf(f) * (1.0f / (1.0f + __expf(-g)));
      zlds[m * 264 + n] = f32_to_bf16_bits(z);
    }
  }

  // ---- stage 2: residual / skip GEMMs, K = 256, LDS-staged weights ----
  v8f accr[8], accs[8];
#pragma unroll
  for (int ct = 0; ct < 8; ++ct) { accr[ct] = acc_zero(); accs[ct] = acc_zero(); }

  const u16* zrow = zlds + (size_t)(m0 + lrow) * 264;
  stage_issue_pair(WrT, WsT, 256, 0, &bstage[0][0][0], &bstage[0][1][0], tid);
#pragma unroll
  for (int kc = 0; kc < 8; ++kc) {
    const int s = kc & 1;
    if (kc < 7)
      stage_issue_pair(WrT, WsT, 256, kc + 1,
                       &bstage[s ^ 1][0][0], &bstage[s ^ 1][1][0], tid);
    if (kc < 7) wait_async8(); else wait_async0();
    __syncthreads();                              // also orders zlds (kc==0)

    v16bf a = frag_ld(zrow, kc * 32, kk);
#pragma unroll
    for (int ct = 0; ct < 8; ++ct) {
      const int n = colBase + ct * 16 + lrow;
      accr[ct] = wmma_bf16(a, frag_ld(&bstage[s][0][(size_t)n * BSTRIDE], 0, kk), accr[ct]);
      accs[ct] = wmma_bf16(a, frag_ld(&bstage[s][1][(size_t)n * BSTRIDE], 0, kk), accs[ct]);
    }
    __syncthreads();
  }

  // ---- writeback: f32 residual chain + bf16 shadow + skip accumulate ----
#pragma unroll
  for (int ct = 0; ct < 8; ++ct) {
    const int n = colBase + ct * 16 + lrow;
    const float brn = br[n], bsn = bs[n];
#pragma unroll
    for (int j = 0; j < 8; ++j) {
      const int m = m0 + khalf * 8 + j;
      const size_t idx = (size_t)(tile0 + m) * F_ + n;
      const float ov = out_in_f32[idx] + accr[ct][j] + brn;
      out_out_f32[idx] = ov;
      out_out_bf[idx]  = f32_to_bf16_bits(ov);
      skip[idx]       += accs[ct][j] + bsn;
    }
  }
}

// ---- head: selu(skip) @ Wd1 + bd1 -> selu -> @ Wd2 + bd2 ------------------

__global__ __launch_bounds__(256) void head_kernel(
    const float* __restrict__ skip,     // [BT][256]
    const u16*   __restrict__ Wd1T,     // [64][256]
    const float* __restrict__ bd1,      // [64]
    const u16*   __restrict__ Wd2T,     // [64][64]
    const float* __restrict__ bd2,      // [64]
    float* __restrict__ out)            // [BT][64]
{
  __shared__ u16 h1[64 * 72];           // padded stride: 72 halves
  const int tid = threadIdx.x;
  const int w = tid >> 5, lane = tid & 31;
  const int lrow = lane & 15, khalf = lane >> 4, kk = khalf * 8;
  const int m0 = (w >> 1) * 16;
  const int ctb = (w & 1) * 2;          // two 16-col tiles per wave (N = 64)
  const int tile0 = blockIdx.x * 64;
  const int rowG = tile0 + m0 + lrow;
  const float* srow = skip + (size_t)rowG * F_;

  // GEMM1: selu(skip)[64x256] @ Wd1T^T -> h1[64x64]
  v8f acc[2];
#pragma unroll
  for (int j2 = 0; j2 < 2; ++j2) acc[j2] = acc_zero();

#pragma unroll
  for (int kc = 0; kc < 8; ++kc) {
    const int kb = kc * 32;
    v16bf a = combine16(selu8_bf(srow + kb + kk), selu8_bf(srow + kb + 16 + kk));
#pragma unroll
    for (int j2 = 0; j2 < 2; ++j2) {
      const int n = (ctb + j2) * 16 + lrow;
      acc[j2] = wmma_bf16(a, frag_ld(Wd1T + (size_t)n * 256, kb, kk), acc[j2]);
    }
  }

#pragma unroll
  for (int j2 = 0; j2 < 2; ++j2) {
    const int n = (ctb + j2) * 16 + lrow;
    const float b1 = bd1[n];
#pragma unroll
    for (int j = 0; j < 8; ++j) {
      const int m = m0 + khalf * 8 + j;
      h1[m * 72 + n] = f32_to_bf16_bits(selu_f(acc[j2][j] + b1));
    }
  }
  __syncthreads();

  // GEMM2: h1[64x64] @ Wd2T^T -> out[64x64]
  v8f acc2[2];
#pragma unroll
  for (int j2 = 0; j2 < 2; ++j2) acc2[j2] = acc_zero();

  const u16* hrow = h1 + (size_t)(m0 + lrow) * 72;
#pragma unroll
  for (int kc = 0; kc < 2; ++kc) {
    const int kb = kc * 32;
    v16bf a = frag_ld(hrow, kb, kk);
#pragma unroll
    for (int j2 = 0; j2 < 2; ++j2) {
      const int n = (ctb + j2) * 16 + lrow;
      acc2[j2] = wmma_bf16(a, frag_ld(Wd2T + (size_t)n * 64, kb, kk), acc2[j2]);
    }
  }

#pragma unroll
  for (int j2 = 0; j2 < 2; ++j2) {
    const int n = (ctb + j2) * 16 + lrow;
    const float b2 = bd2[n];
#pragma unroll
    for (int j = 0; j < 8; ++j) {
      const int m = m0 + khalf * 8 + j;
      out[(size_t)(tile0 + m) * C_ + n] = acc2[j2][j] + b2;
    }
  }
}

// ---------------------------------------------------------------------------

extern "C" void kernel_launch(void* const* d_in, const int* in_sizes, int n_in,
                              void* d_out, int out_size, void* d_ws, size_t ws_size,
                              hipStream_t stream) {
  (void)in_sizes; (void)n_in; (void)out_size; (void)ws_size;

  const float* x   = (const float*)d_in[0];
  const float* Wc  = (const float*)d_in[1];
  const float* Wf  = (const float*)d_in[2];
  const float* Wg  = (const float*)d_in[3];
  const float* Wr  = (const float*)d_in[4];
  const float* br  = (const float*)d_in[5];
  const float* Ws  = (const float*)d_in[6];
  const float* bs  = (const float*)d_in[7];
  const float* Wd1 = (const float*)d_in[8];
  const float* bd1 = (const float*)d_in[9];
  const float* Wd2 = (const float*)d_in[10];
  const float* bd2 = (const float*)d_in[11];
  float* out = (float*)d_out;

  char* ws = (char*)d_ws;
  size_t off = 0;
  auto alloc = [&](size_t bytes) -> char* {
    char* p = ws + off;
    off += (bytes + 255) & ~(size_t)255;
    return p;
  };

  u16*   x_bf   = (u16*)  alloc((size_t)BT_ * C_ * 2);
  float* outAf  = (float*)alloc((size_t)BT_ * F_ * 4);
  float* outBf  = (float*)alloc((size_t)BT_ * F_ * 4);
  u16*   outAb  = (u16*)  alloc((size_t)BT_ * F_ * 2);
  u16*   outBb  = (u16*)  alloc((size_t)BT_ * F_ * 2);
  float* skipb  = (float*)alloc((size_t)BT_ * F_ * 4);
  u16*   WcT    = (u16*)  alloc(256 * 128 * 2);
  u16*   WfT    = (u16*)  alloc((size_t)8 * 256 * 512 * 2);
  u16*   WgT    = (u16*)  alloc((size_t)8 * 256 * 512 * 2);
  u16*   WrT    = (u16*)  alloc((size_t)8 * 256 * 256 * 2);
  u16*   WsT    = (u16*)  alloc((size_t)8 * 256 * 256 * 2);
  u16*   Wd1T   = (u16*)  alloc(64 * 256 * 2);
  u16*   Wd2T   = (u16*)  alloc(64 * 64 * 2);

  // --- prep: convert + transpose weights, convert input ---
  cvt_bf16_kernel<<<(BT_ * C_ + 255) / 256, 256, 0, stream>>>(x, x_bf, BT_ * C_);
  pack_wc_kernel <<<(256 * 128 + 255) / 256,       256, 0, stream>>>(Wc,  WcT);
  pack_wfg_kernel<<<(8 * 256 * 512 + 255) / 256,   256, 0, stream>>>(Wf,  WfT);
  pack_wfg_kernel<<<(8 * 256 * 512 + 255) / 256,   256, 0, stream>>>(Wg,  WgT);
  pack_wrs_kernel<<<(8 * 256 * 256 + 255) / 256,   256, 0, stream>>>(Wr,  WrT);
  pack_wrs_kernel<<<(8 * 256 * 256 + 255) / 256,   256, 0, stream>>>(Ws,  WsT);
  pack_wd1_kernel<<<(64 * 256 + 255) / 256,        256, 0, stream>>>(Wd1, Wd1T);
  pack_wd2_kernel<<<(64 * 64 + 255) / 256,         256, 0, stream>>>(Wd2, Wd2T);

  const int nTiles = BT_ / 64;   // 512 workgroups, 256 threads each

  // --- initial conv (also zeroes skip) ---
  conv_in_kernel<<<nTiles, 256, 0, stream>>>(x_bf, WcT, outAf, outAb, skipb);

  // --- 8 residual layers, ping-pong activation buffers ---
  for (int i = 0; i < 8; ++i) {
    const int d = 2 << i;                       // 2^(i+1)
    const bool even = (i & 1) == 0;
    const float* inF = even ? outAf : outBf;
    const u16*   inB = even ? outAb : outBb;
    float*       oF  = even ? outBf : outAf;
    u16*         oB  = even ? outBb : outAb;
    layer_kernel<<<nTiles, 256, 0, stream>>>(
        inF, inB, oF, oB, skipb,
        WfT + (size_t)i * 256 * 512, WgT + (size_t)i * 256 * 512,
        WrT + (size_t)i * 256 * 256, WsT + (size_t)i * 256 * 256,
        br + i * 256, bs + i * 256, d);
  }

  // --- head ---
  head_kernel<<<nTiles, 256, 0, stream>>>(skipb, Wd1T, bd1, Wd2T, bd2, out);
}
